// AttnBlock_54425825575460
// MI455X (gfx1250) — matrix operand order repaired and checked by reading, not compiled, not run on hardware
//
#include <hip/hip_runtime.h>
#include <hip/hip_bf16.h>

// ---- CDNA5 WMMA types --------------------------------------------------
typedef __attribute__((ext_vector_type(16))) __bf16 v16bf;
typedef __attribute__((ext_vector_type(8)))  float  v8f;

union BFrag { v16bf v; uint4 q[2]; };

static __device__ __forceinline__ float bf2f(unsigned short u) {
  union { unsigned int u; float f; } cv; cv.u = ((unsigned int)u) << 16; return cv.f;
}
static __device__ __forceinline__ unsigned short f2bf(float f) {
  union { float f; unsigned int u; } cv; cv.f = f;
  unsigned int u = cv.u;
  u += 0x7FFFu + ((u >> 16) & 1u);          // round to nearest even
  return (unsigned short)(u >> 16);
}

#define BATCH 8
#define CCH   512
#define NSP   4096            // 64*64
#define NGRP  32
#define CPG   16              // channels per group
#define NHEAD 8
#define HD    64

// ======================================================================
// 1) GroupNorm statistics: one block per (b,g); 16*4096 contiguous floats
// ======================================================================
__global__ __launch_bounds__(256) void gn_stats(const float* __restrict__ x,
                                                float* __restrict__ mean,
                                                float* __restrict__ rstd) {
  const int grp = blockIdx.x;                 // b*32+g ; base = grp*65536 floats
  const float* p = x + ((size_t)grp << 16);
  float s = 0.f, sq = 0.f;
  for (int i = threadIdx.x * 4; i < 65536; i += 256 * 4) {
    float4 v = *(const float4*)(p + i);
    s  += v.x + v.y + v.z + v.w;
    sq += v.x*v.x + v.y*v.y + v.z*v.z + v.w*v.w;
  }
  __shared__ float rs[256], rq[256];
  rs[threadIdx.x] = s; rq[threadIdx.x] = sq;
  __syncthreads();
  for (int o = 128; o > 0; o >>= 1) {
    if (threadIdx.x < o) { rs[threadIdx.x] += rs[threadIdx.x + o];
                           rq[threadIdx.x] += rq[threadIdx.x + o]; }
    __syncthreads();
  }
  if (threadIdx.x == 0) {
    float m = rs[0] * (1.0f / 65536.0f);
    float v = rq[0] * (1.0f / 65536.0f) - m * m;
    mean[grp] = m;
    rstd[grp] = rsqrtf(v + 1e-6f);
  }
}

// ======================================================================
// 2) Apply GN + transpose (b,c,n)f32 -> (b*n, c)bf16 via 32x32 LDS tile
// ======================================================================
__global__ __launch_bounds__(256) void gn_apply_t(const float* __restrict__ x,
                                                  const float* __restrict__ gamma,
                                                  const float* __restrict__ beta,
                                                  const float* __restrict__ mean,
                                                  const float* __restrict__ rstd,
                                                  unsigned short* __restrict__ xT) {
  __shared__ float tile[32][33];
  const int b  = blockIdx.z;
  const int c0 = blockIdx.y * 32;
  const int n0 = blockIdx.x * 32;
  const int tx = threadIdx.x, ty = threadIdx.y;   // (32, 8)
#pragma unroll
  for (int k = 0; k < 4; ++k) {
    int c = c0 + ty + 8 * k;
    float rs = rstd[b * NGRP + (c >> 4)];
    float g  = gamma[c] * rs;
    float mu = mean[b * NGRP + (c >> 4)];
    float be = beta[c];
    float v  = x[(((size_t)(b * CCH + c)) << 12) + n0 + tx];
    tile[ty + 8 * k][tx] = (v - mu) * g + be;
  }
  __syncthreads();
#pragma unroll
  for (int k = 0; k < 4; ++k) {
    int n = n0 + ty + 8 * k;
    xT[((size_t)(b * NSP + n)) * CCH + c0 + tx] = f2bf(tile[tx][ty + 8 * k]);
  }
}

// ======================================================================
// 3) Weights f32 -> bf16 (Wq|Wk|Wv packed, Wp separate)
// ======================================================================
__global__ __launch_bounds__(256) void convert_w(const float* __restrict__ Wq,
                                                 const float* __restrict__ Wk,
                                                 const float* __restrict__ Wv,
                                                 const float* __restrict__ Wp,
                                                 unsigned short* __restrict__ Wb3,
                                                 unsigned short* __restrict__ Wpb) {
  int i = blockIdx.x * 256 + threadIdx.x;            // 0..262143
  Wb3[i]              = f2bf(Wq[i]);
  Wb3[262144 + i]     = f2bf(Wk[i]);
  Wb3[2 * 262144 + i] = f2bf(Wv[i]);
  Wpb[i]              = f2bf(Wp[i]);
}

// ======================================================================
// 4) QKV GEMM: (32768x512)bf16 @ W^T -> (b,c,n)bf16 ; wave = 16M x 64N
// ======================================================================
__global__ __launch_bounds__(32) void gemm_qkv(const unsigned short* __restrict__ A,
                                               const unsigned short* __restrict__ Wb3,
                                               unsigned short* __restrict__ Qo,
                                               unsigned short* __restrict__ Ko,
                                               unsigned short* __restrict__ Vo) {
  const int lane = threadIdx.x;
  const int lo   = lane & 15;      // A row / B col / D col
  const int half = lane >> 4;
  const int M0   = blockIdx.x * 16;         // global row (b*4096+n)
  const int N0   = blockIdx.y * 64;         // output channel base
  const unsigned short* W = Wb3 + (size_t)blockIdx.z * (CCH * CCH);
  unsigned short* Out = (blockIdx.z == 0) ? Qo : ((blockIdx.z == 1) ? Ko : Vo);

  v8f acc[4] = {};
  const unsigned short* arow = A + (size_t)(M0 + lo) * CCH;
  for (int k0 = 0; k0 < CCH; k0 += 32) {
    BFrag a;
    a.q[0] = *(const uint4*)(arow + k0 + 16 * half);
    a.q[1] = *(const uint4*)(arow + k0 + 16 * half + 8);
#pragma unroll
    for (int nb = 0; nb < 4; ++nb) {
      const unsigned short* brow = W + (size_t)(N0 + nb * 16 + lo) * CCH + k0 + 16 * half;
      BFrag bf;
      bf.q[0] = *(const uint4*)(brow);
      bf.q[1] = *(const uint4*)(brow + 8);
      acc[nb] = __builtin_amdgcn_wmma_f32_16x16x32_bf16(
          false, a.v, false, bf.v, (short)0, acc[nb], false, false);
    }
  }
  const int b     = M0 >> 12;
  const int nbase = (M0 & 4095) + 8 * half;   // 8 contiguous n per lane
#pragma unroll
  for (int nb = 0; nb < 4; ++nb) {
    int c = N0 + nb * 16 + lo;
    unsigned short tmp[8];
#pragma unroll
    for (int r = 0; r < 8; ++r) tmp[r] = f2bf(acc[nb][r]);
    *(uint4*)(Out + (((size_t)(b * CCH + c)) << 12) + nbase) = *(const uint4*)tmp;
  }
}

// ======================================================================
// 5) dots[b,ij,n] = sqrt(hd) * sum_d Q[i,d]K[j,d]  (lane = position n)
// ======================================================================
__global__ __launch_bounds__(256) void dots_kernel(const unsigned short* __restrict__ Q,
                                                   const unsigned short* __restrict__ K,
                                                   float* __restrict__ dots) {
  const int b = blockIdx.y;
  const int n = blockIdx.x * 256 + threadIdx.x;
  const size_t base = ((size_t)(b * CCH)) << 12;
  float acc[64];
#pragma unroll
  for (int ij = 0; ij < 64; ++ij) acc[ij] = 0.f;
  for (int d = 0; d < HD; ++d) {
    float q[8], k[8];
#pragma unroll
    for (int i = 0; i < 8; ++i) q[i] = bf2f(Q[base + (((size_t)(i * HD + d)) << 12) + n]);
#pragma unroll
    for (int j = 0; j < 8; ++j) k[j] = bf2f(K[base + (((size_t)(j * HD + d)) << 12) + n]);
#pragma unroll
    for (int i = 0; i < 8; ++i)
#pragma unroll
      for (int j = 0; j < 8; ++j) acc[i * 8 + j] += q[i] * k[j];
  }
  float* drow = dots + (((size_t)(b * 64)) << 12) + n;
#pragma unroll
  for (int ij = 0; ij < 64; ++ij) drow[((size_t)ij) << 12] = acc[ij] * 8.0f;  // * sqrt(64)
}

// ======================================================================
// 6) softmax stats over spatial axis: one block per (b,ij) row of 4096
// ======================================================================
__global__ __launch_bounds__(256) void softmax_stats(const float* __restrict__ dots,
                                                     float* __restrict__ smax,
                                                     float* __restrict__ sinv) {
  const int row = blockIdx.x;                 // b*64 + ij
  const float* p = dots + (((size_t)row) << 12);
  __shared__ float red[256];
  float m = -3.4e38f;
  for (int i = threadIdx.x; i < NSP; i += 256) m = fmaxf(m, p[i]);
  red[threadIdx.x] = m;
  __syncthreads();
  for (int o = 128; o > 0; o >>= 1) {
    if (threadIdx.x < o) red[threadIdx.x] = fmaxf(red[threadIdx.x], red[threadIdx.x + o]);
    __syncthreads();
  }
  const float mx = red[0];
  __syncthreads();
  float s = 0.f;
  for (int i = threadIdx.x; i < NSP; i += 256) s += __expf(p[i] - mx);
  red[threadIdx.x] = s;
  __syncthreads();
  for (int o = 128; o > 0; o >>= 1) {
    if (threadIdx.x < o) red[threadIdx.x] += red[threadIdx.x + o];
    __syncthreads();
  }
  if (threadIdx.x == 0) { smax[row] = mx; sinv[row] = 1.0f / red[0]; }
}

// ======================================================================
// 7) att x V per position -> attout (b*n, c) bf16  (lane = position n)
// ======================================================================
__global__ __launch_bounds__(256) void attv_kernel(const float* __restrict__ dots,
                                                   const float* __restrict__ smax,
                                                   const float* __restrict__ sinv,
                                                   const unsigned short* __restrict__ V,
                                                   unsigned short* __restrict__ attout) {
  const int b = blockIdx.y;
  const int n = blockIdx.x * 256 + threadIdx.x;
  const float* drow = dots + (((size_t)(b * 64)) << 12) + n;
  float att[64];
#pragma unroll
  for (int ij = 0; ij < 64; ++ij)
    att[ij] = __expf(drow[((size_t)ij) << 12] - smax[b * 64 + ij]) * sinv[b * 64 + ij];

  const size_t vb = ((size_t)(b * CCH)) << 12;
  unsigned short* orow = attout + ((size_t)(b * NSP + n)) * CCH;
  for (int dch = 0; dch < 8; ++dch) {
    float v[8][8];
#pragma unroll
    for (int j = 0; j < 8; ++j)
#pragma unroll
      for (int dd = 0; dd < 8; ++dd)
        v[j][dd] = bf2f(V[vb + (((size_t)(j * HD + dch * 8 + dd)) << 12) + n]);
#pragma unroll
    for (int i = 0; i < 8; ++i) {
      unsigned short tmp[8];
#pragma unroll
      for (int dd = 0; dd < 8; ++dd) {
        float o = 0.f;
#pragma unroll
        for (int j = 0; j < 8; ++j) o += att[i * 8 + j] * v[j][dd];
        tmp[dd] = f2bf(o);
      }
      *(uint4*)(orow + i * HD + dch * 8) = *(const uint4*)tmp;
    }
  }
}

// ======================================================================
// 8) Projection GEMM + bias + recomputed-GN residual -> out (b,c,n) f32
// ======================================================================
__global__ __launch_bounds__(32) void gemm_proj(const unsigned short* __restrict__ A,
                                                const unsigned short* __restrict__ Wpb,
                                                const float* __restrict__ bp,
                                                const float* __restrict__ x,
                                                const float* __restrict__ gamma,
                                                const float* __restrict__ beta,
                                                const float* __restrict__ mean,
                                                const float* __restrict__ rstd,
                                                float* __restrict__ out) {
  const int lane = threadIdx.x;
  const int lo   = lane & 15;
  const int half = lane >> 4;
  const int M0   = blockIdx.x * 16;
  const int N0   = blockIdx.y * 64;

  v8f acc[4] = {};
  const unsigned short* arow = A + (size_t)(M0 + lo) * CCH;
  for (int k0 = 0; k0 < CCH; k0 += 32) {
    BFrag a;
    a.q[0] = *(const uint4*)(arow + k0 + 16 * half);
    a.q[1] = *(const uint4*)(arow + k0 + 16 * half + 8);
#pragma unroll
    for (int nb = 0; nb < 4; ++nb) {
      const unsigned short* brow = Wpb + (size_t)(N0 + nb * 16 + lo) * CCH + k0 + 16 * half;
      BFrag bf;
      bf.q[0] = *(const uint4*)(brow);
      bf.q[1] = *(const uint4*)(brow + 8);
      acc[nb] = __builtin_amdgcn_wmma_f32_16x16x32_bf16(
          false, a.v, false, bf.v, (short)0, acc[nb], false, false);
    }
  }
  const int b     = M0 >> 12;
  const int nbase = (M0 & 4095) + 8 * half;
#pragma unroll
  for (int nb = 0; nb < 4; ++nb) {
    int c = N0 + nb * 16 + lo;
    float bias = bp[c];
    float rs   = rstd[b * NGRP + (c >> 4)];
    float mu   = mean[b * NGRP + (c >> 4)];
    float g    = gamma[c] * rs;
    float be   = beta[c];
    const float* xrow = x + (((size_t)(b * CCH + c)) << 12) + nbase;
    float4 x0 = *(const float4*)xrow;
    float4 x1 = *(const float4*)(xrow + 4);
    float res[8] = {x0.x, x0.y, x0.z, x0.w, x1.x, x1.y, x1.z, x1.w};
    float o[8];
#pragma unroll
    for (int r = 0; r < 8; ++r)
      o[r] = acc[nb][r] + bias + (res[r] - mu) * g + be;
    float* orow = out + (((size_t)(b * CCH + c)) << 12) + nbase;
    *(float4*)orow       = make_float4(o[0], o[1], o[2], o[3]);
    *(float4*)(orow + 4) = make_float4(o[4], o[5], o[6], o[7]);
  }
}

// ======================================================================
extern "C" void kernel_launch(void* const* d_in, const int* in_sizes, int n_in,
                              void* d_out, int out_size, void* d_ws, size_t ws_size,
                              hipStream_t stream) {
  (void)in_sizes; (void)n_in; (void)out_size; (void)ws_size;
  const float* x     = (const float*)d_in[0];
  const float* gamma = (const float*)d_in[1];
  const float* beta  = (const float*)d_in[2];
  const float* Wq    = (const float*)d_in[3];
  const float* Wk    = (const float*)d_in[4];
  const float* Wv    = (const float*)d_in[5];
  const float* Wp    = (const float*)d_in[6];
  const float* bp    = (const float*)d_in[7];
  float* out = (float*)d_out;

  char* ws = (char*)d_ws;
  size_t off = 0;
  auto carve = [&](size_t bytes) {
    char* p = ws + off;
    off += (bytes + 255) & ~(size_t)255;
    return p;
  };
  float*          mean   = (float*)carve(256 * sizeof(float));
  float*          rstd   = (float*)carve(256 * sizeof(float));
  float*          smax   = (float*)carve(512 * sizeof(float));
  float*          sinv   = (float*)carve(512 * sizeof(float));
  unsigned short* Wb3    = (unsigned short*)carve((size_t)3 * 512 * 512 * 2);
  unsigned short* Wpb    = (unsigned short*)carve((size_t)512 * 512 * 2);
  unsigned short* xT     = (unsigned short*)carve((size_t)BATCH * NSP * CCH * 2);
  unsigned short* Qb     = (unsigned short*)carve((size_t)BATCH * NSP * CCH * 2);
  unsigned short* Kb     = (unsigned short*)carve((size_t)BATCH * NSP * CCH * 2);
  unsigned short* Vb     = (unsigned short*)carve((size_t)BATCH * NSP * CCH * 2);
  unsigned short* attout = (unsigned short*)carve((size_t)BATCH * NSP * CCH * 2);
  float*          dots   = (float*)carve((size_t)BATCH * 64 * NSP * sizeof(float));

  gn_stats<<<256, 256, 0, stream>>>(x, mean, rstd);
  gn_apply_t<<<dim3(NSP / 32, CCH / 32, BATCH), dim3(32, 8), 0, stream>>>(
      x, gamma, beta, mean, rstd, xT);
  convert_w<<<1024, 256, 0, stream>>>(Wq, Wk, Wv, Wp, Wb3, Wpb);
  gemm_qkv<<<dim3(BATCH * NSP / 16, CCH / 64, 3), 32, 0, stream>>>(xT, Wb3, Qb, Kb, Vb);
  dots_kernel<<<dim3(NSP / 256, BATCH), 256, 0, stream>>>(Qb, Kb, dots);
  softmax_stats<<<BATCH * 64, 256, 0, stream>>>(dots, smax, sinv);
  attv_kernel<<<dim3(NSP / 256, BATCH), 256, 0, stream>>>(dots, smax, sinv, Vb, attout);
  gemm_proj<<<dim3(BATCH * NSP / 16, CCH / 64), 32, 0, stream>>>(
      attout, Wpb, bp, x, gamma, beta, mean, rstd, out);
}
